// DlrmSmall_89455578841757
// MI455X (gfx1250) — matrix operand order repaired and checked by reading, not compiled, not run on hardware
//
#include <hip/hip_runtime.h>

typedef __attribute__((ext_vector_type(16))) _Float16 v16h;
typedef __attribute__((ext_vector_type(8)))  float    v8f;
typedef __attribute__((ext_vector_type(4)))  int      v4i;
typedef unsigned int u32;

#define BATCH_  16384
#define NTAB    26
#define VPT     100000

#if defined(__has_builtin)
#if __has_builtin(__builtin_amdgcn_global_load_async_to_lds_b128) && \
    __has_builtin(__builtin_amdgcn_s_wait_asynccnt)
#define HAVE_ASYNC_LDS 1
#endif
#endif

#if defined(HAVE_ASYNC_LDS)
// Param types per hipcc diagnostic: int4* in addrspace(1) (global) and the
// LDS destination in addrspace(3).
typedef __attribute__((address_space(1))) v4i* gb128_t;
typedef __attribute__((address_space(3))) v4i* lb128_t;
#endif

union Frag { v16h v; u32 u[8]; };

__device__ __forceinline__ v8f wmma_f16(v16h a, v16h b, v8f c) {
  // D = A(16x32 f16) * B(32x16 f16) + C(16x16 f32)
  return __builtin_amdgcn_wmma_f32_16x16x32_f16(false, a, false, b, (short)0, c,
                                                false, false);
}

// ---------------------------------------------------------------------------
// Weight prep: fp32 W[K][N] -> f16 Wt[N][Kpad] (transposed, zero-padded K)
// ---------------------------------------------------------------------------
__global__ void prepw(const float* __restrict__ w, _Float16* __restrict__ wt,
                      int K, int N, int Kpad) {
  int id = blockIdx.x * 256 + threadIdx.x;
  if (id >= N * Kpad) return;
  int n = id / Kpad, k = id - n * Kpad;
  wt[id] = (k < K) ? (_Float16)w[(size_t)k * N + n] : (_Float16)0.f;
}

// ---------------------------------------------------------------------------
// Bottom layer 0: dense(16384x13) @ w0(13x512) + b0, ReLU, f16 out. K=13 is
// too small for WMMA; VALU kernel, weights broadcast via L1.
// ---------------------------------------------------------------------------
__global__ __launch_bounds__(256)
void bot0(const float* __restrict__ dense, const float* __restrict__ w0,
          const float* __restrict__ b0, _Float16* __restrict__ h0) {
  int id = blockIdx.x * 256 + threadIdx.x;
  int m = id >> 9, n = id & 511;
  const float* dr = dense + (size_t)m * 13;
  float s = b0[n];
#pragma unroll
  for (int k = 0; k < 13; ++k) s += dr[k] * w0[k * 512 + n];
  s = s > 0.f ? s : 0.f;
  h0[(size_t)m * 512 + n] = (_Float16)s;
}

// ---------------------------------------------------------------------------
// f16 GEMM + bias + ReLU: C[M][N] = relu(A[M][K] * W[K][N] + bias)
//   A: f16 row-major, lda = K.   Wt: f16 TRANSPOSED weights [N][K].
//   Block tile 64(M) x 128(N), K-step 64. 8 waves: wave -> 16M x 64N.
//   Double-buffered LDS + async global->LDS staging: next tile's 6 async
//   loads/wave are in flight while the current tile's 8 WMMAs execute;
//   in-order ASYNCcnt completion makes s_wait_asynccnt(6) publish exactly
//   the current tile. LDS row stride 72 halves: 16B-aligned rows (B128
//   staging legal), dword-stride 36 -> 16 consecutive rows, 16 banks.
// ---------------------------------------------------------------------------
__global__ __launch_bounds__(256)
void gemm_relu_f16(const _Float16* __restrict__ A, const _Float16* __restrict__ Wt,
                   const float* __restrict__ bias, _Float16* __restrict__ C,
                   int N, int K) {
  __shared__ __align__(16) _Float16 As[2][64][72];
  __shared__ __align__(16) _Float16 Bs[2][128][72];

  const int tid  = threadIdx.x;
  const int lane = tid & 31, wave = tid >> 5;
  const int mt = wave >> 1, nq = wave & 1;
  const int m0 = blockIdx.x * 64, n0 = blockIdx.y * 128;
  const int h = lane >> 4, l16 = lane & 15;

  const v8f vzero = {0.f, 0.f, 0.f, 0.f, 0.f, 0.f, 0.f, 0.f};
  v8f acc[4];
#pragma unroll
  for (int j = 0; j < 4; ++j) acc[j] = vzero;

  const int arow = tid >> 2, acg = (tid & 3) * 16;  // A: 16 halves/thread
  const int brow = tid >> 1, bcg = (tid & 1) * 32;  // B: 32 halves/thread

  auto stage = [&](int buf, int k0) {
    const _Float16* ga = A  + (size_t)(m0 + arow) * K + k0 + acg;
    const _Float16* gb = Wt + (size_t)(n0 + brow) * K + k0 + bcg;
#if defined(HAVE_ASYNC_LDS)
#pragma unroll
    for (int q = 0; q < 2; ++q)
      __builtin_amdgcn_global_load_async_to_lds_b128(
          (gb128_t)(ga + 8 * q), (lb128_t)&As[buf][arow][acg + 8 * q], 0, 0);
#pragma unroll
    for (int q = 0; q < 4; ++q)
      __builtin_amdgcn_global_load_async_to_lds_b128(
          (gb128_t)(gb + 8 * q), (lb128_t)&Bs[buf][brow][bcg + 8 * q], 0, 0);
#else
#pragma unroll
    for (int q = 0; q < 2; ++q)
      *(uint4*)&As[buf][arow][acg + 8 * q] = *(const uint4*)(ga + 8 * q);
#pragma unroll
    for (int q = 0; q < 4; ++q)
      *(uint4*)&Bs[buf][brow][bcg + 8 * q] = *(const uint4*)(gb + 8 * q);
#endif
  };

  const int nk = K >> 6;
  stage(0, 0);

  for (int i = 0; i < nk; ++i) {
    const int cur = i & 1;
    if (i + 1 < nk) stage(cur ^ 1, (i + 1) << 6);
#if defined(HAVE_ASYNC_LDS)
    // In-order completion: <=6 outstanding means current tile's loads done,
    // next tile's 6 still in flight behind the compute below.
    if (i + 1 < nk) __builtin_amdgcn_s_wait_asynccnt(6);
    else            __builtin_amdgcn_s_wait_asynccnt(0);
#endif
    __syncthreads();   // publish buf[cur]

#pragma unroll
    for (int kk = 0; kk < 2; ++kk) {
      const int kb = kk * 32;
      // A fragment (16x32, ISA layout: lanes 0-15 M rows, K pairs per VGPR)
      Frag a;
      const int ra = mt * 16 + l16;
#pragma unroll
      for (int v = 0; v < 4; ++v)
        a.u[v] = *(const u32*)&As[cur][ra][kb + 8 * h + 2 * v];
#pragma unroll
      for (int v = 4; v < 8; ++v)
        a.u[v] = *(const u32*)&As[cur][ra][kb + 16 + 8 * h + 2 * (v - 4)];

      // Load all four B fragments, then issue the WMMAs back-to-back.
      Frag b[4];
#pragma unroll
      for (int j = 0; j < 4; ++j) {
        const int rb = nq * 64 + j * 16 + l16;
#pragma unroll
        for (int v = 0; v < 8; ++v)
          b[j].u[v] = *(const u32*)&Bs[cur][rb][kb + 16 * (v >> 2) + 8 * h + 2 * (v & 3)];
      }
#pragma unroll
      for (int j = 0; j < 4; ++j)
        acc[j] = wmma_f16(a.v, b[j].v, acc[j]);
    }
    __syncthreads();   // all waves done reading buf[cur]; it may be restaged
  }

#pragma unroll
  for (int j = 0; j < 4; ++j) {
    const int n = n0 + nq * 64 + j * 16 + l16;
    const float bv = bias[n];
    const int mb = m0 + mt * 16 + 8 * h;   // C layout: M = r + 8*(lane/16)
#pragma unroll
    for (int r = 0; r < 8; ++r) {
      float x = acc[j][r] + bv;
      x = x > 0.f ? x : 0.f;
      C[(size_t)(mb + r) * N + n] = (_Float16)x;
    }
  }
}

// ---------------------------------------------------------------------------
// Interaction: one wave per sample. Stage 27x128 stack (bot_out + 26 emb
// rows) as f16 in LDS, compute gram S*S^T with 12 WMMAs (upper-tri tiles),
// emit triu(378) + bot_out copy into padded 512-wide top input.
// ---------------------------------------------------------------------------
__global__ __launch_bounds__(128)
void interact(const int* __restrict__ cat, const float* __restrict__ emb,
              const _Float16* __restrict__ bot, _Float16* __restrict__ tin) {
  __shared__ __align__(16) _Float16 S[4][32][132];
  const int lane = threadIdx.x & 31, wid = threadIdx.x >> 5;
  const int sample = blockIdx.x * 4 + wid;
  const int h = lane >> 4, l16 = lane & 15;

  // row 0 = bottom-MLP output (already f16): 4 halves per lane
  uint2 bv = ((const uint2*)(bot + (size_t)sample * 128))[lane];
  *(uint2*)&S[wid][0][lane * 4] = bv;

  // rows 1..26 = embedding rows (f32 gather -> f16). 218 MB total stream.
  for (int t = 0; t < NTAB; ++t) {
    const int idx = cat[sample * NTAB + t] + t * VPT;
    const float4 e = ((const float4*)(emb + (size_t)idx * 128))[lane];
    union { _Float16 hh[4]; uint2 u; } cv;
    cv.hh[0] = (_Float16)e.x; cv.hh[1] = (_Float16)e.y;
    cv.hh[2] = (_Float16)e.z; cv.hh[3] = (_Float16)e.w;
    *(uint2*)&S[wid][1 + t][lane * 4] = cv.u;
  }
  uint2 z; z.x = 0u; z.y = 0u;
#pragma unroll
  for (int r = 27; r < 32; ++r) *(uint2*)&S[wid][r][lane * 4] = z;
  __syncthreads();

  _Float16* trow = tin + (size_t)sample * 512;
  *(uint2*)&trow[lane * 4] = bv;                   // cols [0,128) = bot_out
  if (lane < 6) trow[506 + lane] = (_Float16)0.f;  // zero pad [506,512)

  const int tis[3] = {0, 0, 1};
  const int tjs[3] = {0, 1, 1};
#pragma unroll
  for (int p = 0; p < 3; ++p) {
    const int ti = tis[p], tj = tjs[p];
    v8f c = {0.f, 0.f, 0.f, 0.f, 0.f, 0.f, 0.f, 0.f};
    const int ra = ti * 16 + l16, rb = tj * 16 + l16;
#pragma unroll
    for (int kk = 0; kk < 4; ++kk) {
      const int k0 = kk * 32;
      Frag a, b;
#pragma unroll
      for (int v = 0; v < 4; ++v)
        a.u[v] = *(const u32*)&S[wid][ra][k0 + 8 * h + 2 * v];
#pragma unroll
      for (int v = 4; v < 8; ++v)
        a.u[v] = *(const u32*)&S[wid][ra][k0 + 16 + 8 * h + 2 * (v - 4)];
#pragma unroll
      for (int v = 0; v < 8; ++v)
        b.u[v] = *(const u32*)&S[wid][rb][k0 + 16 * (v >> 2) + 8 * h + 2 * (v & 3)];  // B = S^T
      c = wmma_f16(a.v, b.v, c);
    }
    const int j = tj * 16 + l16;
#pragma unroll
    for (int r = 0; r < 8; ++r) {
      const int i = ti * 16 + r + 8 * h;
      if (i < 27 && j < 27 && i <= j) {
        const int t = i * 27 - (i * (i - 1)) / 2 + (j - i);  // np.triu order
        trow[128 + t] = (_Float16)c[r];
      }
    }
  }
}

// ---------------------------------------------------------------------------
// Final layer: 256 -> 1 dot product, one wave per sample, fp32 output.
// ---------------------------------------------------------------------------
__global__ __launch_bounds__(256)
void top4(const _Float16* __restrict__ t3, const float* __restrict__ w4,
          const float* __restrict__ b4, float* __restrict__ out) {
  const int lane = threadIdx.x & 31, wid = threadIdx.x >> 5;
  const int sample = blockIdx.x * 8 + wid;
  const _Float16* row = t3 + (size_t)sample * 256;
  float s = 0.f;
#pragma unroll
  for (int q = 0; q < 8; ++q)
    s += (float)row[lane * 8 + q] * w4[lane * 8 + q];
#pragma unroll
  for (int off = 16; off; off >>= 1) s += __shfl_xor(s, off, 32);
  if (lane == 0) out[sample] = s + b4[0];
}

// ---------------------------------------------------------------------------
extern "C" void kernel_launch(void* const* d_in, const int* in_sizes, int n_in,
                              void* d_out, int out_size, void* d_ws, size_t ws_size,
                              hipStream_t stream) {
  (void)in_sizes; (void)n_in; (void)out_size; (void)ws_size;
  const float* dense = (const float*)d_in[0];
  const int*   cat   = (const int*)d_in[1];
  const float* emb   = (const float*)d_in[2];
  const float* bw0 = (const float*)d_in[3];  const float* bb0 = (const float*)d_in[4];
  const float* bw1 = (const float*)d_in[5];  const float* bb1 = (const float*)d_in[6];
  const float* bw2 = (const float*)d_in[7];  const float* bb2 = (const float*)d_in[8];
  const float* tw0 = (const float*)d_in[9];  const float* tb0 = (const float*)d_in[10];
  const float* tw1 = (const float*)d_in[11]; const float* tb1 = (const float*)d_in[12];
  const float* tw2 = (const float*)d_in[13]; const float* tb2 = (const float*)d_in[14];
  const float* tw3 = (const float*)d_in[15]; const float* tb3 = (const float*)d_in[16];
  const float* tw4 = (const float*)d_in[17]; const float* tb4 = (const float*)d_in[18];
  float* out = (float*)d_out;

  char* ws = (char*)d_ws;
  size_t off = 0;
  auto take = [&](size_t bytes) -> void* {
    void* p = ws + off;
    off += (bytes + 255) & ~(size_t)255;
    return p;
  };
  _Float16* wt_b1 = (_Float16*)take((size_t)256  * 512  * 2);
  _Float16* wt_b2 = (_Float16*)take((size_t)128  * 256  * 2);
  _Float16* wt_t0 = (_Float16*)take((size_t)1024 * 512  * 2);  // K padded 506->512
  _Float16* wt_t1 = (_Float16*)take((size_t)1024 * 1024 * 2);
  _Float16* wt_t2 = (_Float16*)take((size_t)512  * 1024 * 2);
  _Float16* wt_t3 = (_Float16*)take((size_t)256  * 512  * 2);
  _Float16* bufA  = (_Float16*)take((size_t)BATCH_ * 1024 * 2);  // h0 / t0 / t2
  _Float16* bufB  = (_Float16*)take((size_t)BATCH_ * 1024 * 2);  // h1 / t1 / t3
  _Float16* botb  = (_Float16*)take((size_t)BATCH_ * 128  * 2);
  _Float16* tinb  = (_Float16*)take((size_t)BATCH_ * 512  * 2);

  auto prep = [&](const float* w, _Float16* wt, int K, int N, int Kpad) {
    int elems = N * Kpad;
    prepw<<<(elems + 255) / 256, 256, 0, stream>>>(w, wt, K, N, Kpad);
  };
  prep(bw1, wt_b1, 512, 256, 512);
  prep(bw2, wt_b2, 256, 128, 256);
  prep(tw0, wt_t0, 506, 1024, 512);
  prep(tw1, wt_t1, 1024, 1024, 1024);
  prep(tw2, wt_t2, 1024, 512, 1024);
  prep(tw3, wt_t3, 512, 256, 512);

  bot0<<<(BATCH_ * 512) / 256, 256, 0, stream>>>(dense, bw0, bb0, bufA);

  gemm_relu_f16<<<dim3(BATCH_ / 64, 256 / 128), 256, 0, stream>>>(
      bufA, wt_b1, bb1, bufB, 256, 512);
  gemm_relu_f16<<<dim3(BATCH_ / 64, 128 / 128), 256, 0, stream>>>(
      bufB, wt_b2, bb2, botb, 128, 256);

  interact<<<BATCH_ / 4, 128, 0, stream>>>(cat, emb, botb, tinb);

  gemm_relu_f16<<<dim3(BATCH_ / 64, 1024 / 128), 256, 0, stream>>>(
      tinb, wt_t0, tb0, bufA, 1024, 512);
  gemm_relu_f16<<<dim3(BATCH_ / 64, 1024 / 128), 256, 0, stream>>>(
      bufA, wt_t1, tb1, bufB, 1024, 1024);
  gemm_relu_f16<<<dim3(BATCH_ / 64, 512 / 128), 256, 0, stream>>>(
      bufB, wt_t2, tb2, bufA, 512, 1024);
  gemm_relu_f16<<<dim3(BATCH_ / 64, 256 / 128), 256, 0, stream>>>(
      bufA, wt_t3, tb3, bufB, 256, 512);

  top4<<<BATCH_ / 8, 256, 0, stream>>>(bufB, tw4, tb4, out);
}
